// WanSelfAttention_4638564679736
// MI455X (gfx1250) — matrix operand order repaired and checked by reading, not compiled, not run on hardware
//
#include <hip/hip_runtime.h>
#include <hip/hip_bf16.h>
#include <stdint.h>

#define B_SZ  2
#define S_LEN 2048
#define DIM   2048
#define HEADS 16
#define HD    128
#define EPSV  1e-6f

typedef __attribute__((ext_vector_type(16))) __bf16 v16bf;
typedef __attribute__((ext_vector_type(8)))  float  v8f;
typedef int v4i_vs __attribute__((vector_size(16)));   // matches builtin param type

union BF16x16 {
  v16bf v;
  uint4 q[2];
};

#if defined(__HIP_DEVICE_COMPILE__) && \
    __has_builtin(__builtin_amdgcn_global_load_async_to_lds_b128)
#define HAVE_ASYNC 1
#else
#define HAVE_ASYNC 0
#endif

__device__ __host__ inline __bf16 f2bf(float f) {
  union { float f; uint32_t u; } c; c.f = f;
  uint32_t r = c.u + 0x7FFFu + ((c.u >> 16) & 1u);   // round-to-nearest-even
  union { uint16_t u; __bf16 b; } o; o.u = (uint16_t)(r >> 16);
  return o.b;
}

__device__ inline v8f wmma_bf16(v16bf a, v16bf b, v8f c) {
  return __builtin_amdgcn_wmma_f32_16x16x32_bf16(false, a, false, b, (short)0, c,
                                                 false, false);
}

__device__ inline float rmax16(float x) {
#pragma unroll
  for (int m = 1; m <= 8; m <<= 1) x = fmaxf(x, __shfl_xor(x, m, 32));
  return x;
}
__device__ inline float rsum16(float x) {
#pragma unroll
  for (int m = 1; m <= 8; m <<= 1) x += __shfl_xor(x, m, 32);
  return x;
}

// 16B global -> LDS copy: async (ASYNCcnt) when available, sync fallback.
__device__ __forceinline__ void copy16_g2l(const __bf16* g, __bf16* l) {
#if HAVE_ASYNC
  __builtin_amdgcn_global_load_async_to_lds_b128(
      (__attribute__((address_space(1))) v4i_vs*)g,
      (__attribute__((address_space(3))) v4i_vs*)l, 0, 0);
#else
  *(uint4*)l = *(const uint4*)g;
#endif
}

__device__ __forceinline__ void wait_async0() {
#if HAVE_ASYNC
#if __has_builtin(__builtin_amdgcn_s_wait_asynccnt)
  __builtin_amdgcn_s_wait_asynccnt(0);
#else
  asm volatile("s_wait_asynccnt 0x0" ::: "memory");
#endif
#endif
}

// ---------------------------------------------------------------- converts
__global__ void cvt_bf16(const float* __restrict__ in, __bf16* __restrict__ out, int n) {
  int i = blockIdx.x * blockDim.x + threadIdx.x;
  int stride = gridDim.x * blockDim.x;
  for (; i < n; i += stride) out[i] = f2bf(in[i]);
}

// ---------------------------------------------------------------- block GEMM
// C[128x128 tile] = A[128xK] * W[128xK]^T. 8 waves, each computing a 64x32
// register tile (4x2 WMMA tiles). A/B k-panels (128x32 bf16, 8KB each) are
// double-buffered in LDS and filled with async global->LDS copies.
__device__ __forceinline__ void stage_panels(
    const __bf16* __restrict__ gA, const __bf16* __restrict__ gB,
    __bf16* sA, __bf16* sB, int tid, int kk) {
#pragma unroll
  for (int p = 0; p < 2; ++p) {
    int cid = tid + p * 256;            // 512 chunks of 16B per 8KB panel
    int row = cid >> 2;
    int c   = (cid & 3) * 8;            // bf16 elements
    copy16_g2l(gA + (size_t)row * DIM + kk + c, sA + row * 32 + c);
    copy16_g2l(gB + (size_t)row * DIM + kk + c, sB + row * 32 + c);
  }
}

template <bool BIAS>
__device__ __forceinline__ void gemm_block(
    const __bf16* __restrict__ A, const __bf16* __restrict__ W,
    const float* __restrict__ bias, float* __restrict__ C, int m0, int n0) {
  __shared__ __align__(16) __bf16 sA[2][128 * 32];
  __shared__ __align__(16) __bf16 sB[2][128 * 32];

  const int tid  = threadIdx.x;
  const int wave = tid >> 5, lane = tid & 31;
  const int half = lane >> 4, lrow = lane & 15;
  const int wm = wave >> 2;            // 0..1 : 64-row strip
  const int wn = wave & 3;             // 0..3 : 32-col strip

  const __bf16* gA = A + (size_t)m0 * DIM;
  const __bf16* gB = W + (size_t)n0 * DIM;

  v8f acc[4][2];
#pragma unroll
  for (int mt = 0; mt < 4; ++mt)
#pragma unroll
    for (int nt = 0; nt < 2; ++nt) acc[mt][nt] = (v8f){};

  stage_panels(gA, gB, sA[0], sB[0], tid, 0);

  const int KS = DIM / 32;
  for (int ks = 0; ks < KS; ++ks) {
    const int cur = ks & 1;
    wait_async0();        // this wave's writes to buf[cur] have landed
    __syncthreads();      // all waves' writes landed; buf[cur^1] free to refill
    if (ks + 1 < KS) stage_panels(gA, gB, sA[cur ^ 1], sB[cur ^ 1], tid, (ks + 1) * 32);

    BF16x16 bfr[2];
#pragma unroll
    for (int nt = 0; nt < 2; ++nt) {
      const __bf16* p = &sB[cur][(wn * 32 + nt * 16 + lrow) * 32 + half * 8];
      bfr[nt].q[0] = *(const uint4*)(p);
      bfr[nt].q[1] = *(const uint4*)(p + 16);
    }
#pragma unroll
    for (int mt = 0; mt < 4; ++mt) {
      BF16x16 afr;
      const __bf16* p = &sA[cur][(wm * 64 + mt * 16 + lrow) * 32 + half * 8];
      afr.q[0] = *(const uint4*)(p);
      afr.q[1] = *(const uint4*)(p + 16);
#pragma unroll
      for (int nt = 0; nt < 2; ++nt)
        acc[mt][nt] = wmma_bf16(afr.v, bfr[nt].v, acc[mt][nt]);
    }
  }

#pragma unroll
  for (int mt = 0; mt < 4; ++mt)
#pragma unroll
    for (int nt = 0; nt < 2; ++nt) {
      int row0 = m0 + wm * 64 + mt * 16 + half * 8;
      int col  = n0 + wn * 32 + nt * 16 + lrow;
      float badd = BIAS ? bias[col] : 0.0f;
      float* crow = C + (size_t)row0 * DIM + col;
#pragma unroll
      for (int v2 = 0; v2 < 8; ++v2) crow[(size_t)v2 * DIM] = acc[mt][nt][v2] + badd;
    }
}

__global__ __launch_bounds__(256) void gemm_qkv_blk(
    const __bf16* __restrict__ xb,
    const __bf16* __restrict__ wq, const __bf16* __restrict__ wk,
    const __bf16* __restrict__ wv,
    float* __restrict__ qf, float* __restrict__ kf, float* __restrict__ vf) {
  int bid = blockIdx.x;                 // 3 * 32 * 16 blocks
  int which = bid >> 9;
  int t = bid & 511;
  int bm = t >> 4, bn = t & 15;
  const __bf16* W = (which == 0) ? wq : (which == 1) ? wk : wv;
  float*        C = (which == 0) ? qf : (which == 1) ? kf : vf;
  gemm_block<false>(xb, W, nullptr, C, bm * 128, bn * 128);
}

__global__ __launch_bounds__(256) void gemm_out_blk(
    const __bf16* __restrict__ ab, const __bf16* __restrict__ wo,
    const float* __restrict__ bo, float* __restrict__ out) {
  int t = blockIdx.x;                   // 32 * 16 blocks
  int bm = t >> 4, bn = t & 15;
  gemm_block<true>(ab, wo, bo, out, bm * 128, bn * 128);
}

// ------------------------------------------------- RMSNorm + RoPE + pack/transpose
__global__ __launch_bounds__(256) void pack_kernel(
    const float* __restrict__ qf, const float* __restrict__ kf,
    const float* __restrict__ vf,
    const float* __restrict__ gq, const float* __restrict__ gk,
    const float* __restrict__ freqs,
    __bf16* __restrict__ qb, __bf16* __restrict__ kb, __bf16* __restrict__ vT) {
  const int tok = blockIdx.x;          // b*S + pos
  const int b   = tok >> 11;
  const int pos = tok & (S_LEN - 1);

  const float* q = qf + (size_t)tok * DIM;
  const float* k = kf + (size_t)tok * DIM;
  const float* v = vf + (size_t)tok * DIM;
  const int base = threadIdx.x * 8;

  float qv[8], kv[8];
  float sq = 0.f, sk = 0.f;
#pragma unroll
  for (int e = 0; e < 8; ++e) {
    qv[e] = q[base + e]; kv[e] = k[base + e];
    sq += qv[e] * qv[e]; sk += kv[e] * kv[e];
  }
  __shared__ float redq[256], redk[256];
  redq[threadIdx.x] = sq; redk[threadIdx.x] = sk;
  __syncthreads();
  for (int st = 128; st > 0; st >>= 1) {
    if (threadIdx.x < st) {
      redq[threadIdx.x] += redq[threadIdx.x + st];
      redk[threadIdx.x] += redk[threadIdx.x + st];
    }
    __syncthreads();
  }
  const float rq = rsqrtf(redq[0] * (1.0f / DIM) + EPSV);
  const float rk = rsqrtf(redk[0] * (1.0f / DIM) + EPSV);

  const float* fr = freqs + (size_t)pos * (HD / 2);
  __bf16* qo = qb + (size_t)tok * DIM;
  __bf16* ko = kb + (size_t)tok * DIM;
#pragma unroll
  for (int p = 0; p < 4; ++p) {
    int i0 = base + 2 * p;
    int fi = (i0 & (HD - 1)) >> 1;
    float ang = fr[fi];
    float cs = __cosf(ang), sn = __sinf(ang);
    float q0 = qv[2 * p] * rq * gq[i0], q1 = qv[2 * p + 1] * rq * gq[i0 + 1];
    float k0 = kv[2 * p] * rk * gk[i0], k1 = kv[2 * p + 1] * rk * gk[i0 + 1];
    qo[i0]     = f2bf(q0 * cs - q1 * sn);
    qo[i0 + 1] = f2bf(q0 * sn + q1 * cs);
    ko[i0]     = f2bf(k0 * cs - k1 * sn);
    ko[i0 + 1] = f2bf(k0 * sn + k1 * cs);
  }
#pragma unroll
  for (int e = 0; e < 8; ++e) {
    int i = base + e;
    int h = i >> 7, d = i & (HD - 1);
    vT[(((size_t)b * HEADS + h) * HD + d) * S_LEN + pos] = f2bf(v[i]);
  }
}

// ---------------------------------------------------------------- flash attention
// One wave per (batch, head, 16-query tile). Online softmax over 32-key blocks.
__global__ __launch_bounds__(256) void attn_kernel(
    const __bf16* __restrict__ qbp, const __bf16* __restrict__ kbp,
    const __bf16* __restrict__ vT, const int* __restrict__ seq_lens,
    __bf16* __restrict__ attn_b) {
  __shared__ __align__(16) __bf16 Pst[8][16 * 32];   // per-wave P staging

  const int wave = threadIdx.x >> 5;
  const int lane = threadIdx.x & 31;
  const int half = lane >> 4;
  const int lrow = lane & 15;

  int gw = blockIdx.x * 8 + wave;          // b*H*(S/16) = 4096 waves
  const int qt = gw & 127;
  const int bh = gw >> 7;
  const int h  = bh & (HEADS - 1);
  const int b  = bh >> 4;
  const int m0 = qt * 16;
  const int slen = seq_lens[b];

  const float sm_scale = 0.08838834764831845f;   // 1/sqrt(128)

  // Q fragments: 4 K-chunks of 32 (head dim 128)
  BF16x16 qf4[4];
  const __bf16* qrow = qbp + (((size_t)b * S_LEN + (m0 + lrow)) * HEADS + h) * HD;
#pragma unroll
  for (int c = 0; c < 4; ++c) {
    qf4[c].q[0] = *(const uint4*)(qrow + c * 32 + half * 8);
    qf4[c].q[1] = *(const uint4*)(qrow + c * 32 + half * 8 + 16);
  }

  float m_run[8], l_run[8];
  v8f o[8];
#pragma unroll
  for (int v2 = 0; v2 < 8; ++v2) { m_run[v2] = -1e30f; l_run[v2] = 0.f; }
#pragma unroll
  for (int t = 0; t < 8; ++t) o[t] = (v8f){};

  for (int j = 0; j < S_LEN / 32; ++j) {
    const int kb0 = j * 32;
    v8f s0 = {}, s1 = {};
    const __bf16* krow0 =
        kbp + (((size_t)b * S_LEN + kb0 + lrow) * HEADS + h) * HD + half * 8;
    const __bf16* krow1 = krow0 + (size_t)16 * HEADS * HD;
#pragma unroll
    for (int c = 0; c < 4; ++c) {
      BF16x16 b0, b1;
      b0.q[0] = *(const uint4*)(krow0 + c * 32);
      b0.q[1] = *(const uint4*)(krow0 + c * 32 + 16);
      b1.q[0] = *(const uint4*)(krow1 + c * 32);
      b1.q[1] = *(const uint4*)(krow1 + c * 32 + 16);
      s0 = wmma_bf16(qf4[c].v, b0.v, s0);
      s1 = wmma_bf16(qf4[c].v, b1.v, s1);
    }

    const bool ok0 = (kb0 + lrow) < slen;
    const bool ok1 = (kb0 + 16 + lrow) < slen;
#pragma unroll
    for (int v2 = 0; v2 < 8; ++v2) {
      float a0 = ok0 ? s0[v2] * sm_scale : -1e30f;
      float a1 = ok1 ? s1[v2] * sm_scale : -1e30f;
      float mt = rmax16(fmaxf(a0, a1));
      float mn = fmaxf(m_run[v2], mt);
      float alpha = __expf(m_run[v2] - mn);
      float p0 = __expf(a0 - mn);
      float p1 = __expf(a1 - mn);
      float rs = rsum16(p0 + p1);
      l_run[v2] = l_run[v2] * alpha + rs;
      m_run[v2] = mn;
#pragma unroll
      for (int t = 0; t < 8; ++t) o[t][v2] *= alpha;
      int r = half * 8 + v2;
      Pst[wave][r * 32 + lrow]      = f2bf(p0);
      Pst[wave][r * 32 + lrow + 16] = f2bf(p1);
    }

    // reload P in A-matrix layout (16x32 bf16)
    BF16x16 pf;
    const __bf16* pl = &Pst[wave][lrow * 32 + half * 8];
    pf.q[0] = *(const uint4*)(pl);
    pf.q[1] = *(const uint4*)(pl + 16);

#pragma unroll
    for (int t = 0; t < 8; ++t) {
      BF16x16 vfgr;
      const __bf16* vr =
          vT + (((size_t)b * HEADS + h) * HD + t * 16 + lrow) * S_LEN + kb0 + half * 8;
      vfgr.q[0] = *(const uint4*)(vr);
      vfgr.q[1] = *(const uint4*)(vr + 16);
      o[t] = wmma_bf16(pf.v, vfgr.v, o[t]);
    }
  }

  __bf16* orow = attn_b + ((size_t)b * S_LEN + m0) * DIM + h * HD;
#pragma unroll
  for (int v2 = 0; v2 < 8; ++v2) {
    float inv = 1.0f / l_run[v2];
    int r = half * 8 + v2;
#pragma unroll
    for (int t = 0; t < 8; ++t)
      orow[(size_t)r * DIM + t * 16 + lrow] = f2bf(o[t][v2] * inv);
  }
}

// ---------------------------------------------------------------- launch
extern "C" void kernel_launch(void* const* d_in, const int* in_sizes, int n_in,
                              void* d_out, int out_size, void* d_ws, size_t ws_size,
                              hipStream_t stream) {
  const float* x        = (const float*)d_in[0];
  const int*   seq_lens = (const int*)d_in[1];
  // d_in[2] grid_sizes: unused by the reference math
  const float* freqs = (const float*)d_in[3];
  const float* Wq    = (const float*)d_in[4];
  const float* Wk    = (const float*)d_in[5];
  const float* Wv    = (const float*)d_in[6];
  const float* Wo    = (const float*)d_in[7];
  const float* bo    = (const float*)d_in[8];
  const float* gq    = (const float*)d_in[9];
  const float* gk    = (const float*)d_in[10];
  float* out = (float*)d_out;

  char* ws = (char*)d_ws;
  size_t off = 0;
  auto alloc = [&](size_t bytes) -> void* {
    void* p = ws + off;
    off += (bytes + 255) & ~(size_t)255;
    return p;
  };
  const size_t NX = (size_t)B_SZ * S_LEN * DIM;   // 8.4M elems
  const size_t NW = (size_t)DIM * DIM;            // 4.2M elems

  __bf16* xb  = (__bf16*)alloc(NX * 2);
  __bf16* wqb = (__bf16*)alloc(NW * 2);
  __bf16* wkb = (__bf16*)alloc(NW * 2);
  __bf16* wvb = (__bf16*)alloc(NW * 2);
  __bf16* wob = (__bf16*)alloc(NW * 2);
  float*  qf  = (float*)alloc(NX * 4);
  float*  kf  = (float*)alloc(NX * 4);
  float*  vf  = (float*)alloc(NX * 4);
  __bf16* qb  = (__bf16*)alloc(NX * 2);
  __bf16* kb  = (__bf16*)alloc(NX * 2);
  __bf16* vT  = (__bf16*)alloc(NX * 2);
  __bf16* attn_b = (__bf16*)qf;   // qf is dead after pack_kernel

  cvt_bf16<<<dim3(2048), 256, 0, stream>>>(x,  xb,  (int)NX);
  cvt_bf16<<<dim3(1024), 256, 0, stream>>>(Wq, wqb, (int)NW);
  cvt_bf16<<<dim3(1024), 256, 0, stream>>>(Wk, wkb, (int)NW);
  cvt_bf16<<<dim3(1024), 256, 0, stream>>>(Wv, wvb, (int)NW);
  cvt_bf16<<<dim3(1024), 256, 0, stream>>>(Wo, wob, (int)NW);

  gemm_qkv_blk<<<dim3(3 * 32 * 16), 256, 0, stream>>>(xb, wqb, wkb, wvb, qf, kf, vf);
  pack_kernel<<<dim3(B_SZ * S_LEN), 256, 0, stream>>>(qf, kf, vf, gq, gk, freqs,
                                                      qb, kb, vT);
  attn_kernel<<<dim3(B_SZ * HEADS * (S_LEN / 16) / 8), 256, 0, stream>>>(
      qb, kb, vT, seq_lens, attn_b);
  gemm_out_blk<<<dim3(32 * 16), 256, 0, stream>>>(attn_b, wob, bo, out);
}
